// Head_87634512708241
// MI455X (gfx1250) — matrix operand order repaired and checked by reading, not compiled, not run on hardware
//
#include <hip/hip_runtime.h>

#define B_DIM 8
#define T_DIM 2048
#define C_DIM 1024
#define HS    64

typedef __attribute__((ext_vector_type(16))) __bf16 v16bf;
typedef __attribute__((ext_vector_type(8)))  float  v8f;

union ABfrag {
    v16bf v;
    unsigned u[8];
    uint4 q[2];
};

__device__ __forceinline__ v8f wmma_bf16(v16bf a, v16bf b, v8f c) {
    return __builtin_amdgcn_wmma_f32_16x16x32_bf16(
        /*neg_a=*/false, a, /*neg_b=*/false, b,
        /*c_mod=*/(short)0, c, /*reuse_a=*/false, /*reuse_b=*/false);
}

__device__ __forceinline__ unsigned short to_bf16(float f) {
    unsigned u = __builtin_bit_cast(unsigned, f);
    return (unsigned short)((u + 0x8000u) >> 16);    // round-to-nearest
}

// pack two fp32 -> packed bf16 pair: round bias + single v_perm_b32
__device__ __forceinline__ unsigned pack2_bf16(float a, float b) {
    unsigned ua = __builtin_bit_cast(unsigned, a) + 0x8000u;
    unsigned ub = __builtin_bit_cast(unsigned, b) + 0x8000u;
    // bytes: [b.hi16 | a.hi16]  (sel 0-3 = s1, 4-7 = s0)
    return __builtin_amdgcn_perm(ub, ua, 0x07060302u);
}

// butterfly xor across lanes with immediate ds_swizzle (group-of-32 encoding)
template <int MASK>
__device__ __forceinline__ float swz_xor(float v) {
    int i = __builtin_amdgcn_ds_swizzle(__builtin_bit_cast(int, v), (MASK << 10) | 0x1F);
    return __builtin_bit_cast(float, i);
}
__device__ __forceinline__ float rowmax16(float v) {
    v = fmaxf(v, swz_xor<1>(v));
    v = fmaxf(v, swz_xor<2>(v));
    v = fmaxf(v, swz_xor<4>(v));
    v = fmaxf(v, swz_xor<8>(v));
    return v;
}
__device__ __forceinline__ float rowsum16(float v) {
    v += swz_xor<1>(v);
    v += swz_xor<2>(v);
    v += swz_xor<4>(v);
    v += swz_xor<8>(v);
    return v;
}

// ---------------------------------------------------------------------------
// Kernel 1: fp32 -> bf16 weight conversion (Wq/Wk/Wv, 64x1024 each)
// ---------------------------------------------------------------------------
__global__ void convert_w(const float* __restrict__ Wq, const float* __restrict__ Wk,
                          const float* __restrict__ Wv,
                          unsigned short* __restrict__ Wqb, unsigned short* __restrict__ Wkb,
                          unsigned short* __restrict__ Wvb) {
    int i = blockIdx.x * 256 + threadIdx.x;
    const float* src = (blockIdx.y == 0) ? Wq : (blockIdx.y == 1) ? Wk : Wv;
    unsigned short* dst = (blockIdx.y == 0) ? Wqb : (blockIdx.y == 1) ? Wkb : Wvb;
    dst[i] = to_bf16(src[i]);
}

// ---------------------------------------------------------------------------
// Kernel 2: QKV projection.  grid = (256, 3), block = 128 (4 waves).
// Each wave: one 16-row tile of x, produces 16x64 of Q, K, or V.
// Q/K stored row-major bf16 [B*T, 64]; V stored transposed bf16 [B, 64, T].
// ---------------------------------------------------------------------------
__global__ void qkv_proj(const float* __restrict__ x,
                         const unsigned short* __restrict__ Wqb,
                         const unsigned short* __restrict__ Wkb,
                         const unsigned short* __restrict__ Wvb,
                         unsigned short* __restrict__ Qb,
                         unsigned short* __restrict__ Kb,
                         unsigned short* __restrict__ Vtb) {
    const int lane = threadIdx.x & 31;
    const int wave = threadIdx.x >> 5;
    const int half = lane >> 4;     // 0: lanes 0-15, 1: lanes 16-31
    const int l16  = lane & 15;

    const int tile = blockIdx.x * 4 + wave;   // 0 .. B*T/16-1 (=1023)
    const int b    = tile >> 7;               // 128 tiles per batch
    const int trow = (tile & 127) * 16;
    const int mat  = blockIdx.y;              // 0=Q 1=K 2=V

    const unsigned short* W = (mat == 0) ? Wqb : (mat == 1) ? Wkb : Wvb;

    // A-matrix row for this lane (M = lane%16)
    const float* xrow = x + ((size_t)(b * T_DIM + trow + l16)) * C_DIM;
    const unsigned short* wlane = W + (size_t)l16 * C_DIM + half * 8;

    v8f acc[4];
#pragma unroll
    for (int nt = 0; nt < 4; ++nt) acc[nt] = (v8f){0,0,0,0,0,0,0,0};

    for (int cc = 0; cc < C_DIM; cc += 32) {
        // ---- issue ALL loads for this chunk first (one clause, overlap) ----
        const float* xa = xrow + cc + half * 8;
        __builtin_prefetch(xa + 32, 0, 1);                 // next chunk of x
        float4 f0 = *(const float4*)(xa + 0);
        float4 f1 = *(const float4*)(xa + 4);
        float4 f2 = *(const float4*)(xa + 16);
        float4 f3 = *(const float4*)(xa + 20);

        ABfrag Bf[4];
#pragma unroll
        for (int nt = 0; nt < 4; ++nt) {
            const unsigned short* wr = wlane + (size_t)(nt * 16) * C_DIM + cc;
            Bf[nt].q[0] = *(const uint4*)(wr);
            Bf[nt].q[1] = *(const uint4*)(wr + 16);
        }

        // ---- A fragment: x[M, cc..cc+31] -> 16-bit A layout ----
        // low lanes hold K{0..7,16..23}, high lanes K{8..15,24..31}
        ABfrag A;
        A.u[0] = pack2_bf16(f0.x, f0.y);  A.u[1] = pack2_bf16(f0.z, f0.w);
        A.u[2] = pack2_bf16(f1.x, f1.y);  A.u[3] = pack2_bf16(f1.z, f1.w);
        A.u[4] = pack2_bf16(f2.x, f2.y);  A.u[5] = pack2_bf16(f2.z, f2.w);
        A.u[6] = pack2_bf16(f3.x, f3.y);  A.u[7] = pack2_bf16(f3.z, f3.w);

#pragma unroll
        for (int nt = 0; nt < 4; ++nt)
            acc[nt] = wmma_bf16(A.v, Bf[nt].v, acc[nt]);
    }

    // ---- branch-free store: (base, row-stride, col-stride), mat is uniform ----
    unsigned short* dst;
    size_t rs, cs;
    if (mat == 0)      { dst = Qb  + (size_t)b * T_DIM * HS; rs = HS; cs = 1; }
    else if (mat == 1) { dst = Kb  + (size_t)b * T_DIM * HS; rs = HS; cs = 1; }
    else               { dst = Vtb + (size_t)b * HS * T_DIM; rs = 1;  cs = T_DIM; }

    unsigned short* dlane = dst + (size_t)(trow + 8 * half) * rs + (size_t)l16 * cs;
#pragma unroll
    for (int nt = 0; nt < 4; ++nt)
#pragma unroll
        for (int r = 0; r < 8; ++r)
            dlane[(size_t)r * rs + (size_t)(nt * 16) * cs] = to_bf16(acc[nt][r]);
}

// ---------------------------------------------------------------------------
// Kernel 3: flash attention.  grid = (T/128, B), block = 256 (8 waves).
// Each wave owns 16 query rows; loops over 32-key chunks (causal).
// ---------------------------------------------------------------------------
__global__ void attn(const unsigned short* __restrict__ Qb,
                     const unsigned short* __restrict__ Kb,
                     const unsigned short* __restrict__ Vtb,
                     float* __restrict__ out) {
    __shared__ __align__(16) unsigned short Plds[8][16 * 32];  // per-wave P staging (1 KB)

    const int lane = threadIdx.x & 31;
    const int wave = threadIdx.x >> 5;
    const int half = lane >> 4;
    const int l16  = lane & 15;
    const int b    = blockIdx.y;
    const int q0   = blockIdx.x * 128 + wave * 16;

    // ---- load Q as two 16x32 A-fragments (kept in registers all kernel) ----
    ABfrag Af[2];
    {
        const unsigned short* qr = Qb + ((size_t)(b * T_DIM + q0 + l16)) * HS;
#pragma unroll
        for (int c = 0; c < 2; ++c) {
            const unsigned short* p = qr + c * 32 + half * 8;
            Af[c].q[0] = *(const uint4*)(p);
            Af[c].q[1] = *(const uint4*)(p + 16);
        }
    }

    v8f O[4];
    float m[8], l[8];
#pragma unroll
    for (int nt = 0; nt < 4; ++nt) O[nt] = (v8f){0,0,0,0,0,0,0,0};
#pragma unroll
    for (int r = 0; r < 8; ++r) { m[r] = -3.0e38f; l[r] = 0.0f; }

    const float scale = 0.125f;  // HS^-0.5 = 1/8
    const int qlast = q0 + 15;
    unsigned short* pl = &Plds[wave][0];

    const unsigned short* krow0 = Kb  + ((size_t)(b * T_DIM + l16)) * HS + half * 8;
    const unsigned short* vrow0 = Vtb + ((size_t)b * HS + l16) * T_DIM + half * 8;

    for (int s0 = 0; s0 <= qlast; s0 += 32) {
        // prefetch next chunk of K and V (speculative; dropped if OOB)
        __builtin_prefetch(krow0 + (size_t)(s0 + 32) * HS, 0, 1);
        __builtin_prefetch(vrow0 + s0 + 32, 0, 1);

        // ---- issue all 4 K B-fragment loads, then the WMMA quartet ----
        ABfrag Bk[4];   // [st*2 + c]
#pragma unroll
        for (int st = 0; st < 2; ++st) {
            const unsigned short* kr = krow0 + (size_t)(s0 + st * 16) * HS;
#pragma unroll
            for (int c = 0; c < 2; ++c) {
                Bk[st * 2 + c].q[0] = *(const uint4*)(kr + c * 32);
                Bk[st * 2 + c].q[1] = *(const uint4*)(kr + c * 32 + 16);
            }
        }
        v8f S[2];
#pragma unroll
        for (int st = 0; st < 2; ++st) {
            v8f acc = (v8f){0,0,0,0,0,0,0,0};
            acc = wmma_bf16(Af[0].v, Bk[st * 2 + 0].v, acc);
            acc = wmma_bf16(Af[1].v, Bk[st * 2 + 1].v, acc);
            S[st] = acc;
        }

        // ---- scale + causal mask ----
#pragma unroll
        for (int st = 0; st < 2; ++st) {
#pragma unroll
            for (int r = 0; r < 8; ++r) {
                int key  = s0 + st * 16 + l16;
                int qrow = q0 + r + 8 * half;
                float v = S[st][r] * scale;
                S[st][r] = (key <= qrow) ? v : -3.0e38f;
            }
        }

        // ---- online softmax ----
        float alpha[8];
#pragma unroll
        for (int r = 0; r < 8; ++r) {
            float mx = rowmax16(fmaxf(S[0][r], S[1][r]));
            float mnew = fmaxf(m[r], mx);
            alpha[r] = __expf(m[r] - mnew);
            m[r] = mnew;
        }
#pragma unroll
        for (int st = 0; st < 2; ++st)
#pragma unroll
            for (int r = 0; r < 8; ++r)
                S[st][r] = __expf(S[st][r] - m[r]);
#pragma unroll
        for (int r = 0; r < 8; ++r) {
            float sm = rowsum16(S[0][r] + S[1][r]);
            l[r] = l[r] * alpha[r] + sm;
        }
#pragma unroll
        for (int nt = 0; nt < 4; ++nt)
#pragma unroll
            for (int r = 0; r < 8; ++r)
                O[nt][r] *= alpha[r];

        // ---- V B-fragments: issue all 4 loads early (overlap with LDS work) ----
        ABfrag Bv[4];
#pragma unroll
        for (int nt = 0; nt < 4; ++nt) {
            const unsigned short* vp = vrow0 + (size_t)(nt * 16) * T_DIM + s0;
            Bv[nt].q[0] = *(const uint4*)(vp);
            Bv[nt].q[1] = *(const uint4*)(vp + 16);
        }

        // ---- P: C-layout -> A-layout via wave-private LDS (bf16 16x32) ----
#pragma unroll
        for (int r = 0; r < 8; ++r) {
            pl[(r + 8 * half) * 32 + l16]      = to_bf16(S[0][r]);
            pl[(r + 8 * half) * 32 + 16 + l16] = to_bf16(S[1][r]);
        }
        asm volatile("" ::: "memory");  // DS ops are in-order per wave; block compiler reordering
        ABfrag Pf;
        {
            const unsigned short* pr = pl + l16 * 32 + half * 8;
            Pf.q[0] = *(const uint4*)(pr);
            Pf.q[1] = *(const uint4*)(pr + 16);
        }

        // ---- O += P * V ----
#pragma unroll
        for (int nt = 0; nt < 4; ++nt)
            O[nt] = wmma_bf16(Pf.v, Bv[nt].v, O[nt]);
    }

    // ---- epilogue: divide by row sum, store fp32 ----
#pragma unroll
    for (int nt = 0; nt < 4; ++nt)
#pragma unroll
        for (int r = 0; r < 8; ++r) {
            int row = q0 + r + 8 * half;
            out[((size_t)(b * T_DIM + row)) * HS + nt * 16 + l16] = O[nt][r] / l[r];
        }
}

// ---------------------------------------------------------------------------
extern "C" void kernel_launch(void* const* d_in, const int* in_sizes, int n_in,
                              void* d_out, int out_size, void* d_ws, size_t ws_size,
                              hipStream_t stream) {
    const float* x  = (const float*)d_in[0];
    const float* Wk = (const float*)d_in[1];
    const float* Wq = (const float*)d_in[2];
    const float* Wv = (const float*)d_in[3];
    float* out = (float*)d_out;

    unsigned short* ws  = (unsigned short*)d_ws;
    unsigned short* Wqb = ws;                       // 64*1024
    unsigned short* Wkb = Wqb + 64 * 1024;
    unsigned short* Wvb = Wkb + 64 * 1024;
    unsigned short* Qb  = Wvb + 64 * 1024;          // [B*T, 64]
    unsigned short* Kb  = Qb  + (size_t)B_DIM * T_DIM * HS;
    unsigned short* Vtb = Kb  + (size_t)B_DIM * T_DIM * HS;  // [B, 64, T]

    convert_w<<<dim3((64 * 1024) / 256, 3), 256, 0, stream>>>(Wq, Wk, Wv, Wqb, Wkb, Wvb);
    qkv_proj<<<dim3(256, 3), 128, 0, stream>>>(x, Wqb, Wkb, Wvb, Qb, Kb, Vtb);
    attn<<<dim3(T_DIM / 128, B_DIM), 256, 0, stream>>>(Qb, Kb, Vtb, out);
}
